// LTSGNN_34514357190968
// MI455X (gfx1250) — compile-verified, matrix operand order checked by reading, not run on hardware
//
#include <hip/hip_runtime.h>
#include <hip/hip_bf16.h>

#define NN 50000
#define NE 800000
#define NG 16

typedef __attribute__((ext_vector_type(16))) _Float16 v16h;
typedef __attribute__((ext_vector_type(8)))  _Float16 v8h;
typedef __attribute__((ext_vector_type(8)))  float    v8f;

static __host__ __device__ inline int cdivi(long long a, int b) { return (int)((a + b - 1) / b); }

__device__ inline unsigned encf(float f) {
    unsigned u = __float_as_uint(f);
    return (u & 0x80000000u) ? ~u : (u | 0x80000000u);
}
__device__ inline float decf(unsigned u) {
    return __uint_as_float((u & 0x80000000u) ? (u & 0x7fffffffu) : ~u);
}
__device__ inline float lrelu(float x) { return x > 0.f ? x : 0.2f * x; }

// ---------------- generic zero ----------------
__global__ void k_fzero(float* p, long long n) {
    long long t = (long long)blockIdx.x * blockDim.x + threadIdx.x;
    if (t < n) p[t] = 0.f;
}

// ---------------- degree + self-loop attr (mean of incoming edge_attr) ----------------
__global__ void k_deg(const int* __restrict__ src, const int* __restrict__ dst,
                      const float* __restrict__ eattr, float* cnt, float* lattr) {
    int e = blockIdx.x * blockDim.x + threadIdx.x;
    if (e >= NE) return;
    int d = dst[e];
    atomicAdd(cnt + d, 1.f);
    atomicAdd(lattr + 3 * d + 0, eattr[3 * e + 0]);
    atomicAdd(lattr + 3 * d + 1, eattr[3 * e + 1]);
    atomicAdd(lattr + 3 * d + 2, eattr[3 * e + 2]);
}
__global__ void k_loopdiv(const float* __restrict__ cnt, float* lattr) {
    int n = blockIdx.x * blockDim.x + threadIdx.x;
    if (n >= NN) return;
    float c = fmaxf(cnt[n], 1.f);
    lattr[3 * n + 0] /= c; lattr[3 * n + 1] /= c; lattr[3 * n + 2] /= c;
}

// ---------------- f32 -> f16 staging (K zero-padded) ----------------
// Xh[n*KP + k] = (k < Kreal) ? f16(X[n*Kreal + k]) : 0
__global__ void k_cvt_xh(const float* __restrict__ X, _Float16* __restrict__ Xh,
                         int Kreal, int KP, long long total) {
    long long t = (long long)blockIdx.x * blockDim.x + threadIdx.x;
    if (t >= total) return;
    int k = (int)(t % KP);
    long long n = t / KP;
    Xh[t] = (k < Kreal) ? (_Float16)X[n * Kreal + k] : (_Float16)0.f;
}
// Wt[col*KP + k] = (k < Kreal) ? f16(W[k*Ncols + col]) : 0   (transposed)
__global__ void k_cvt_wt(const float* __restrict__ W, _Float16* __restrict__ Wt,
                         int Kreal, int KP, int Ncols) {
    int t = blockIdx.x * blockDim.x + threadIdx.x;
    if (t >= Ncols * KP) return;
    int k = t % KP, col = t / KP;
    Wt[t] = (k < Kreal) ? (_Float16)W[(long long)k * Ncols + col] : (_Float16)0.f;
}

// ---------------- WMMA GEMM: H[N,Ncols] = Xh[N,KP] @ Wt[Ncols,KP]^T ----------------
// One wave per 16x16 tile; wave w of the block = column tile w.
// A frag (16x32 f16): lane -> M = lane%16, kb = (lane<16?0:8);
//   halves 0..7 = K=kk+kb+i (contiguous), halves 8..15 = K=kk+16+kb+i (contiguous)
//   -> two b128 loads per matrix per K-step. B symmetric with N=lane%16 on Wt rows.
// C/D (16x16 f32): lane -> N = lane%16, vgpr r -> M = r + 8*(lane/16).
template <int KP>
__global__ void k_gemm_wmma(const _Float16* __restrict__ Xh, const _Float16* __restrict__ Wt,
                            float* __restrict__ H, int Ncols) {
    const int lane = threadIdx.x & 31;
    const int wave = threadIdx.x >> 5;
    const int rowBase = blockIdx.x << 4;          // NN % 16 == 0
    const int m16 = lane & 15;
    const int kb  = (lane >> 4) << 3;
    const int col = (wave << 4) + m16;
    const _Float16* xrow = Xh + (long long)(rowBase + m16) * KP + kb;
    const _Float16* wrow = Wt + (long long)col * KP + kb;
    v8f acc = {};
#pragma unroll
    for (int kk = 0; kk < KP; kk += 32) {
        v8h a0 = *(const v8h*)(xrow + kk);
        v8h a1 = *(const v8h*)(xrow + kk + 16);
        v8h b0 = *(const v8h*)(wrow + kk);
        v8h b1 = *(const v8h*)(wrow + kk + 16);
        v16h a = __builtin_shufflevector(a0, a1, 0, 1, 2, 3, 4, 5, 6, 7,
                                         8, 9, 10, 11, 12, 13, 14, 15);
        v16h b = __builtin_shufflevector(b0, b1, 0, 1, 2, 3, 4, 5, 6, 7,
                                         8, 9, 10, 11, 12, 13, 14, 15);
        acc = __builtin_amdgcn_wmma_f32_16x16x32_f16(false, a, false, b,
                                                     (short)0, acc, false, false);
    }
    const int hi = lane >> 4;
#pragma unroll
    for (int r = 0; r < 8; ++r) {
        int row = rowBase + (hi << 3) + r;
        H[(long long)row * Ncols + col] = acc[r];
    }
}

// ---------------- wea = We @ ae  (3-vector) ----------------
__global__ void k_wea(const float* __restrict__ We, const float* __restrict__ ae,
                      float* wea, int C) {
    int j = threadIdx.x;
    if (j < 3) {
        float s = 0.f;
        for (int c = 0; c < C; ++c) s += We[j * C + c] * ae[c];
        wea[j] = s;
    }
}

// ---------------- out init with bias ----------------
__global__ void k_out_init(float* o, const float* __restrict__ b, int C, long long n) {
    long long t = (long long)blockIdx.x * blockDim.x + threadIdx.x;
    if (t < n) o[t] = b[t % C];
}

// ---------------- per-node: alpha_s/alpha_d dots, self logit, seed max, zero den ----------
__global__ void k_node_prep(const float* __restrict__ h, const float* __restrict__ a_s,
                            const float* __restrict__ a_d, const float* __restrict__ lattr,
                            const float* __restrict__ wea, int C,
                            float* asb, float* adb, float* slb, unsigned* menc, float* den) {
    int n = blockIdx.x * blockDim.x + threadIdx.x;
    if (n >= NN) return;
    const float4* hv = (const float4*)(h + (long long)n * C);
    const float4* sv = (const float4*)a_s;
    const float4* dv = (const float4*)a_d;
    float as = 0.f, ad = 0.f;
    for (int q = 0; q < (C >> 2); ++q) {
        float4 v = hv[q], s = sv[q], d = dv[q];
        as += v.x * s.x + v.y * s.y + v.z * s.z + v.w * s.w;
        ad += v.x * d.x + v.y * d.y + v.z * d.z + v.w * d.w;
    }
    float ale = lattr[3 * n] * wea[0] + lattr[3 * n + 1] * wea[1] + lattr[3 * n + 2] * wea[2];
    float l = lrelu(as + ad + ale);
    asb[n] = as; adb[n] = ad; slb[n] = l;
    menc[n] = encf(l);   // every node has a self-loop -> valid max seed
    den[n] = 0.f;
}

// ---------------- per-edge logit + segment max ----------------
__global__ void k_edge_logit(const int* __restrict__ src, const int* __restrict__ dst,
                             const float* __restrict__ eattr,
                             const float* __restrict__ asb, const float* __restrict__ adb,
                             const float* __restrict__ wea, float* logE, unsigned* menc) {
    int e = blockIdx.x * blockDim.x + threadIdx.x;
    if (e >= NE) return;
    float al = eattr[3 * e] * wea[0] + eattr[3 * e + 1] * wea[1] + eattr[3 * e + 2] * wea[2];
    float l = lrelu(asb[src[e]] + adb[dst[e]] + al);
    logE[e] = l;
    atomicMax(menc + dst[e], encf(l));
}

// ---------------- exp passes ----------------
__global__ void k_self_exp(const float* __restrict__ slb, const unsigned* __restrict__ menc,
                           float* exS, float* den) {
    int n = blockIdx.x * blockDim.x + threadIdx.x;
    if (n >= NN) return;
    float ex = expf(slb[n] - decf(menc[n]));
    exS[n] = ex;
    atomicAdd(den + n, ex);
}
__global__ void k_edge_exp(const int* __restrict__ dst, const float* __restrict__ logE,
                           const unsigned* __restrict__ menc, float* exE, float* den) {
    int e = blockIdx.x * blockDim.x + threadIdx.x;
    if (e >= NE) return;
    int d = dst[e];
    float ex = expf(logE[e] - decf(menc[d]));
    exE[e] = ex;
    atomicAdd(den + d, ex);
}

// ---------------- weighted scatter: out[dst] += coef * h[src] ----------------
__global__ void k_scatter(const int* __restrict__ src, const int* __restrict__ dst,
                          const float* __restrict__ exE, const float* __restrict__ den,
                          const float* __restrict__ h, float* out, int C) {
    long long t = (long long)blockIdx.x * blockDim.x + threadIdx.x;
    int qpe = C >> 2;
    if (t >= (long long)NE * qpe) return;
    int e = (int)(t / qpe), q = (int)(t % qpe);
    int s = src[e], d = dst[e];
    float coef = exE[e] / den[d];
    float4 v = ((const float4*)(h + (long long)s * C))[q];
    float* o = out + (long long)d * C + (q << 2);
    atomicAdd(o + 0, coef * v.x); atomicAdd(o + 1, coef * v.y);
    atomicAdd(o + 2, coef * v.z); atomicAdd(o + 3, coef * v.w);
}
__global__ void k_self_scatter(const float* __restrict__ exS, const float* __restrict__ den,
                               const float* __restrict__ h, float* out, int C) {
    long long t = (long long)blockIdx.x * blockDim.x + threadIdx.x;
    int qpe = C >> 2;
    if (t >= (long long)NN * qpe) return;
    int n = (int)(t / qpe), q = (int)(t % qpe);
    float coef = exS[n] / den[n];
    float4 v = ((const float4*)(h + (long long)n * C))[q];
    float* o = out + (long long)n * C + (q << 2);
    o[0] += coef * v.x; o[1] += coef * v.y; o[2] += coef * v.z; o[3] += coef * v.w;
}

__global__ void k_relu(float* p, long long n) {
    long long t = (long long)blockIdx.x * blockDim.x + threadIdx.x;
    if (t < n) p[t] = fmaxf(p[t], 0.f);
}

// ---------------- global mean pool ----------------
__global__ void k_pool_acc(const float* __restrict__ h, const int* __restrict__ batch,
                           float* pool, float* pcnt, int C) {
    long long t = (long long)blockIdx.x * blockDim.x + threadIdx.x;
    int qpe = C >> 2;
    if (t >= (long long)NN * qpe) return;
    int n = (int)(t / qpe), q = (int)(t % qpe);
    int g = batch[n];
    float4 v = ((const float4*)(h + (long long)n * C))[q];
    float* p = pool + (long long)g * C + (q << 2);
    atomicAdd(p + 0, v.x); atomicAdd(p + 1, v.y);
    atomicAdd(p + 2, v.z); atomicAdd(p + 3, v.w);
    if (q == 0) atomicAdd(pcnt + g, 1.f);
}
__global__ void k_pool_final(const float* __restrict__ pool, const float* __restrict__ pcnt,
                             float* out, int C) {
    int t = blockIdx.x * blockDim.x + threadIdx.x;
    if (t >= NG * C) return;
    out[t] = pool[t] / fmaxf(pcnt[t / C], 1.f);
}

// ---------------- one GAT layer ----------------
template <int KP>
static void run_layer(hipStream_t stream, const float* X, int Kreal,
                      const float* W, const float* a_s, const float* a_d,
                      const float* We, const float* ae, const float* b, int C,
                      const int* src, const int* dst, const float* eattr,
                      const float* lattr, _Float16* xh, _Float16* wt,
                      float* h, float* o,
                      float* asb, float* adb, float* slb, unsigned* menc,
                      float* den, float* exS, float* logE, float* exE, float* wea) {
    k_cvt_xh<<<cdivi((long long)NN * KP, 256), 256, 0, stream>>>(X, xh, Kreal, KP,
                                                                 (long long)NN * KP);
    k_cvt_wt<<<cdivi((long long)C * KP, 256), 256, 0, stream>>>(W, wt, Kreal, KP, C);
    k_gemm_wmma<KP><<<NN / 16, 32 * (C / 16), 0, stream>>>(xh, wt, h, C);
    k_wea<<<1, 32, 0, stream>>>(We, ae, wea, C);
    k_out_init<<<cdivi((long long)NN * C, 256), 256, 0, stream>>>(o, b, C, (long long)NN * C);
    k_node_prep<<<cdivi(NN, 256), 256, 0, stream>>>(h, a_s, a_d, lattr, wea, C,
                                                    asb, adb, slb, menc, den);
    k_edge_logit<<<cdivi(NE, 256), 256, 0, stream>>>(src, dst, eattr, asb, adb, wea, logE, menc);
    k_self_exp<<<cdivi(NN, 256), 256, 0, stream>>>(slb, menc, exS, den);
    k_edge_exp<<<cdivi(NE, 256), 256, 0, stream>>>(dst, logE, menc, exE, den);
    k_scatter<<<cdivi((long long)NE * (C / 4), 256), 256, 0, stream>>>(src, dst, exE, den, h, o, C);
    k_self_scatter<<<cdivi((long long)NN * (C / 4), 256), 256, 0, stream>>>(exS, den, h, o, C);
}

extern "C" void kernel_launch(void* const* d_in, const int* in_sizes, int n_in,
                              void* d_out, int out_size, void* d_ws, size_t ws_size,
                              hipStream_t stream) {
    const float* x     = (const float*)d_in[0];
    const int*   ei    = (const int*)d_in[1];
    const float* eattr = (const float*)d_in[2];
    const int*   batch = (const int*)d_in[3];
    const float* W1  = (const float*)d_in[4];
    const float* as1 = (const float*)d_in[5];
    const float* ad1 = (const float*)d_in[6];
    const float* We1 = (const float*)d_in[7];
    const float* ae1 = (const float*)d_in[8];
    const float* b1  = (const float*)d_in[9];
    const float* W2  = (const float*)d_in[10];
    const float* as2 = (const float*)d_in[11];
    const float* ad2 = (const float*)d_in[12];
    const float* We2 = (const float*)d_in[13];
    const float* ae2 = (const float*)d_in[14];
    const float* b2  = (const float*)d_in[15];
    float* out = (float*)d_out;
    const int* src = ei;
    const int* dst = ei + NE;

    // carve workspace (256B-aligned slabs)
    char* w = (char*)d_ws;
    auto carve = [&](size_t bytes) {
        void* p = (void*)w;
        w += (bytes + 255) & ~(size_t)255;
        return p;
    };
    float*     hbuf  = (float*)carve((size_t)NN * 128 * 4);
    float*     obuf  = (float*)carve((size_t)NN * 128 * 4);
    _Float16*  xh    = (_Float16*)carve((size_t)NN * 128 * 2);
    _Float16*  wt    = (_Float16*)carve((size_t)128 * 128 * 2);
    float*     cnt   = (float*)carve((size_t)NN * 4);
    float*     lattr = (float*)carve((size_t)NN * 3 * 4);
    float*     asb   = (float*)carve((size_t)NN * 4);
    float*     adb   = (float*)carve((size_t)NN * 4);
    float*     slb   = (float*)carve((size_t)NN * 4);
    unsigned*  menc  = (unsigned*)carve((size_t)NN * 4);
    float*     den   = (float*)carve((size_t)NN * 4);
    float*     exS   = (float*)carve((size_t)NN * 4);
    float*     logE  = (float*)carve((size_t)NE * 4);
    float*     exE   = (float*)carve((size_t)NE * 4);
    float*     wea   = (float*)carve(256);
    float*     pool  = (float*)carve((size_t)NG * 64 * 4);
    float*     pcnt  = (float*)carve((size_t)NG * 4);

    // in-degree + mean self-loop edge_attr (shared by both layers)
    k_fzero<<<cdivi(NN, 256), 256, 0, stream>>>(cnt, NN);
    k_fzero<<<cdivi((long long)NN * 3, 256), 256, 0, stream>>>(lattr, (long long)NN * 3);
    k_deg<<<cdivi(NE, 256), 256, 0, stream>>>(src, dst, eattr, cnt, lattr);
    k_loopdiv<<<cdivi(NN, 256), 256, 0, stream>>>(cnt, lattr);

    // layer 1: X[N,5] -> obuf[N,128]   (K padded 5 -> 32)
    run_layer<32>(stream, x, 5, W1, as1, ad1, We1, ae1, b1, 128,
                  src, dst, eattr, lattr, xh, wt, hbuf, obuf,
                  asb, adb, slb, menc, den, exS, logE, exE, wea);
    k_relu<<<cdivi((long long)NN * 128, 256), 256, 0, stream>>>(obuf, (long long)NN * 128);

    // layer 2: obuf[N,128] -> obuf[N,64]  (gemm consumes staged copy, so obuf reuse is safe)
    run_layer<128>(stream, obuf, 128, W2, as2, ad2, We2, ae2, b2, 64,
                   src, dst, eattr, lattr, xh, wt, hbuf, obuf,
                   asb, adb, slb, menc, den, exS, logE, exE, wea);

    // global mean pool -> d_out [16,64]
    k_fzero<<<cdivi((long long)NG * 64, 256), 256, 0, stream>>>(pool, (long long)NG * 64);
    k_fzero<<<1, 32, 0, stream>>>(pcnt, NG);
    k_pool_acc<<<cdivi((long long)NN * 16, 256), 256, 0, stream>>>(obuf, batch, pool, pcnt, 64);
    k_pool_final<<<cdivi((long long)NG * 64, 256), 256, 0, stream>>>(pool, pcnt, out, 64);
}
